// SHSAttention_18133351924335
// MI455X (gfx1250) — matrix operand order repaired and checked by reading, not compiled, not run on hardware
//
#include <hip/hip_runtime.h>
#include <stdint.h>

#define HIDDEN 768
#define PDIM   164                 // int(768/4.67)
#define X2DIM  (HIDDEN - PDIM)     // 604
#define QKDIM  16
#define QKVOUT (2*QKDIM + PDIM)    // 196
#define BATCH  8
#define SEQ    2048
#define NTOK   (BATCH*SEQ)
#define VPAD   176                 // PDIM padded to multiple of 16
#define NVT    (VPAD/16)           // 11 value tiles
#define LNEPS  1e-5f
#define APAD   776                 // LDS row stride (halves) for A tile, bank-safe
#define KVBLK  (512 + VPAD*32)     // 6144 halves = 12 KB per 32-key block

typedef __attribute__((ext_vector_type(16))) _Float16 v16h;
typedef __attribute__((ext_vector_type(8)))  _Float16 v8h;
typedef __attribute__((ext_vector_type(8)))  float    v8f;

static __device__ __forceinline__ v8f wmma_f16(v16h a, v16h b, v8f c) {
  // emits v_wmma_f32_16x16x32_f16
  return __builtin_amdgcn_wmma_f32_16x16x32_f16(false, a, false, b, (short)0, c,
                                                false, false);
}

// Fused DPP ROW_XMASK max reduction step: one v_max_num_f32_dpp per step
// (dst = max(dpp_xmask(x), x)), mirroring the v_add_f32_dpp form the
// compiler emits for the sum tree.
static __device__ __forceinline__ float halfwave_max(float x) {
  float y;
  asm("v_max_num_f32_dpp %0, %1, %1 row_xmask:1 row_mask:0xf bank_mask:0xf bound_ctrl:1"
      : "=v"(y) : "v"(x));
  asm("v_max_num_f32_dpp %0, %1, %1 row_xmask:2 row_mask:0xf bank_mask:0xf bound_ctrl:1"
      : "=v"(x) : "v"(y));
  asm("v_max_num_f32_dpp %0, %1, %1 row_xmask:4 row_mask:0xf bank_mask:0xf bound_ctrl:1"
      : "=v"(y) : "v"(x));
  asm("v_max_num_f32_dpp %0, %1, %1 row_xmask:8 row_mask:0xf bank_mask:0xf bound_ctrl:1"
      : "=v"(x) : "v"(y));
  return x;
}
// Sum tree: intrinsic form already fuses to v_add_f32_dpp.
template <int M>
static __device__ __forceinline__ float xmask_add(float x) {
  float y = __int_as_float(__builtin_amdgcn_update_dpp(
      0, __float_as_int(x), 0x160 + M, 0xF, 0xF, true));
  return x + y;
}
static __device__ __forceinline__ float halfwave_add(float x) {
  x = xmask_add<1>(x); x = xmask_add<2>(x);
  x = xmask_add<4>(x); x = xmask_add<8>(x);
  return x;
}

// ---------------------------------------------------------------- kernel 0
__global__ void __launch_bounds__(256)
cvt_w_kernel(const float* __restrict__ w, _Float16* __restrict__ wh, int n) {
  int i = blockIdx.x * blockDim.x + threadIdx.x;
  if (i < n) wh[i] = (_Float16)w[i];
}

// ---------------------------------------------------------------- kernel 1
// One wave per token: LayerNorm(x1) + QKV projection.  q is pre-scaled by
// 1/sqrt(QK_DIM).  k/v are written into pre-tiled 32-key KV blocks:
//   block = [K: key*16+d (32x16)][V^T: 512 + d*32+key (176x32, zero-padded)]
// so the attention kernel stages each block with one contiguous 12KB copy.
__global__ void __launch_bounds__(256)
ln_qkv_kernel(const float* __restrict__ hs,
              const float* __restrict__ lng, const float* __restrict__ lnb,
              const float* __restrict__ qw,  const float* __restrict__ qb,
              _Float16* __restrict__ qf, _Float16* __restrict__ kvt,
              _Float16* __restrict__ mg) {
  __shared__ float xs[8][PDIM];
  const int w = threadIdx.x >> 5, lane = threadIdx.x & 31;
  const size_t t = (size_t)blockIdx.x * 8 + w;
  const float* x = hs + t * HIDDEN;

  const int s   = (int)(t & (SEQ - 1));
  const int bi  = (int)(t >> 11);
  const int key = s & 31;
  _Float16* kv = kvt + (size_t)(bi * (SEQ / 32) + (s >> 5)) * KVBLK;

  float sm = 0.f, s2 = 0.f;
  for (int d = lane; d < PDIM; d += 32) { float v = x[d]; sm += v; s2 += v * v; }
  #pragma unroll
  for (int m = 16; m; m >>= 1) { sm += __shfl_xor(sm, m, 32); s2 += __shfl_xor(s2, m, 32); }
  const float mean = sm * (1.f / PDIM);
  const float var  = s2 * (1.f / PDIM) - mean * mean;
  const float rstd = rsqrtf(var + LNEPS);
  for (int d = lane; d < PDIM; d += 32)
    xs[w][d] = (x[d] - mean) * rstd * lng[d] + lnb[d];
  __syncthreads();

  for (int o = lane; o < QKVOUT; o += 32) {
    float a = qb[o];
    const float* wr = qw + (size_t)o * PDIM;
    #pragma unroll 4
    for (int d = 0; d < PDIM; ++d) a += xs[w][d] * wr[d];
    if (o < QKDIM)          qf[t * QKDIM + o]                  = (_Float16)(a * 0.25f);
    else if (o < 2*QKDIM)   kv[key * 16 + (o - QKDIM)]          = (_Float16)a;
    else                    kv[512 + (o - 2*QKDIM) * 32 + key]  = (_Float16)a;
  }
  if (lane < VPAD - PDIM) kv[512 + (PDIM + lane) * 32 + key] = (_Float16)0.f;
  for (int j = lane; j < X2DIM; j += 32)
    mg[t * HIDDEN + PDIM + j] = (_Float16)x[PDIM + j];
}

// ---------------------------------------------------------------- kernel 2
// Flash attention.  Block = 8 waves; wave w owns a 16-query tile.  All waves
// share the batch's K/V stream: each 32-key block is one contiguous 12KB
// record (K tile + transposed V tile) staged with a straight coalesced copy.
// K-pad rows (d=16..31) come from a shared zero row (LDS broadcast).
__global__ void __launch_bounds__(256)
attn_kernel(const _Float16* __restrict__ qf, const _Float16* __restrict__ kvt,
            _Float16* __restrict__ mg) {
  __shared__ __align__(32) _Float16 kvS[KVBLK];         // [0..511]=K [512..]=V^T
  __shared__ __align__(32) _Float16 pS[8][16][32];      // per-wave P tile
  __shared__ __align__(32) _Float16 zS[16];             // zero row (K pad)

  const int tid  = threadIdx.x, w = tid >> 5, lane = tid & 31;
  const int half = lane >> 4,  l16 = lane & 15;
  const int b    = blockIdx.x >> 4;                     // 16 blocks per batch
  const int q0   = (blockIdx.x & 15) * 128 + w * 16;
  const size_t base = (size_t)b * SEQ;

  if (tid < 8) ((uint32_t*)zS)[tid] = 0u;

  // Q fragment (A layout, K padded 16->32 with zeros), loaded once.
  v16h aq;
  {
    const _Float16* qr = qf + (base + q0 + l16) * QKDIM;
    v16h q16 = *(const v16h*)qr;
    #pragma unroll
    for (int i = 0; i < 16; ++i) { aq[i] = half ? (_Float16)0.f : q16[i]; }
  }
  // Hoisted B-fragment LDS pointers; half==1 lanes read the shared zero row.
  const _Float16* kp0 = half ? zS : &kvS[l16 * 16];
  const _Float16* kp1 = half ? zS : &kvS[(16 + l16) * 16];

  v8f acc[NVT] = {};
  float mrow[8], lrow[8];
  #pragma unroll
  for (int j = 0; j < 8; ++j) { mrow[j] = -3.0e38f; lrow[j] = 0.f; }

  for (int kb = 0; kb < SEQ / 32; ++kb) {
    __syncthreads();                                    // tile free to overwrite
    {
      const uint32_t* src = (const uint32_t*)(kvt + (size_t)(b * (SEQ/32) + kb) * KVBLK);
      uint32_t* dst = (uint32_t*)kvS;
      #pragma unroll
      for (int i = 0; i < KVBLK / 2 / 256; ++i)         // 12 coalesced dwords/thread
        dst[tid + i * 256] = src[tid + i * 256];
    }
    __syncthreads();

    // ---- scores S = Q*K^T (two 16-key groups)
    v16h bk0 = *(const v16h*)kp0;
    v16h bk1 = *(const v16h*)kp1;
    v8f c0 = {}, c1 = {};
    c0 = wmma_f16(aq, bk0, c0);
    c1 = wmma_f16(aq, bk1, c1);

    // ---- online softmax; row r = j + 8*half; DPP ROW_XMASK reductions
    #pragma unroll
    for (int j = 0; j < 8; ++j) {
      const float mx   = halfwave_max(fmaxf(c0[j], c1[j]));
      const float mn   = fmaxf(mrow[j], mx);
      const float corr = __expf(mrow[j] - mn);
      const float p0   = __expf(c0[j] - mn);
      const float p1   = __expf(c1[j] - mn);
      const float rs   = halfwave_add(p0 + p1);
      lrow[j] = lrow[j] * corr + rs;
      mrow[j] = mn;
      #pragma unroll
      for (int t = 0; t < NVT; ++t) acc[t][j] = acc[t][j] * corr;
      const int r = j + half * 8;
      pS[w][r][l16]      = (_Float16)p0;                // C-layout -> LDS
      pS[w][r][l16 + 16] = (_Float16)p1;
    }

    // ---- re-load P in A-fragment layout
    v16h ap;
    {
      const _Float16* pr = &pS[w][l16][0];
      v8h lo = *(const v8h*)(pr + half * 16);
      v8h hi = *(const v8h*)(pr + half * 16 + 8);
      #pragma unroll
      for (int i = 0; i < 8; ++i) { ap[i] = lo[i]; ap[i + 8] = hi[i]; }
    }
    // ---- acc += P @ V   (11 WMMAs; B from transposed V tile, contiguous)
    #pragma unroll
    for (int t = 0; t < NVT; ++t) {
      v16h bv = *(const v16h*)&kvS[512 + (t * 16 + l16) * 32 + half * 16];
      acc[t] = wmma_f16(ap, bv, acc[t]);
    }
  }

  // ---- epilogue: attended / l  ->  merged[..., 0:PDIM] as f16
  #pragma unroll
  for (int j = 0; j < 8; ++j) {
    const float inv = 1.f / lrow[j];
    const size_t tok = base + q0 + j + half * 8;
    #pragma unroll
    for (int t = 0; t < NVT; ++t) {
      const int d = t * 16 + l16;
      if (d < PDIM) mg[tok * HIDDEN + d] = (_Float16)(acc[t][j] * inv);
    }
  }
}

// ---------------------------------------------------------------- kernel 3
// out = merged @ out_w^T + out_b.  Block = 12 waves; block owns one 16-row
// m-tile staged ONCE in LDS (A read from memory exactly once); wave w owns
// n-group w (64 cols).  LDS A row stride padded to 776 halves so the 16-lane
// fragment reads hit distinct banks.  No barriers in the k-loop.
__global__ void __launch_bounds__(384)
outproj_kernel(const _Float16* __restrict__ mg, const _Float16* __restrict__ wh,
               const float* __restrict__ ob, float* __restrict__ out) {
  __shared__ __align__(32) _Float16 aS[16 * APAD];      // 16 x 768 (padded) f16

  const int tid  = threadIdx.x;
  const int lane = tid & 31, half = lane >> 4, l16 = lane & 15;
  const int w    = tid >> 5;                            // n-group 0..11
  const size_t m0 = (size_t)blockIdx.x * 16;
  const int n0 = w * 64;

  // ---- stage A tile: 16 rows x 384 dwords, coalesced, no divisions
  {
    const uint32_t* src = (const uint32_t*)(mg + m0 * HIDDEN);
    uint32_t* dst = (uint32_t*)aS;
    #pragma unroll
    for (int r = 0; r < 16; ++r)
      dst[r * (APAD / 2) + tid] = src[r * (HIDDEN / 2) + tid];
  }
  __syncthreads();

  v8f acc[4] = {};
  const _Float16* arow = &aS[l16 * APAD];

  for (int k = 0; k < HIDDEN; k += 32) {
    v16h a;
    {
      const _Float16* ar = arow + k + half * 16;        // K {0-15}/{16-31}
      v8h lo = *(const v8h*)ar;
      v8h hi = *(const v8h*)(ar + 8);
      #pragma unroll
      for (int i = 0; i < 8; ++i) { a[i] = lo[i]; a[i + 8] = hi[i]; }
    }
    #pragma unroll
    for (int t = 0; t < 4; ++t) {
      // B fragment: column n = lane, 16 contiguous K halves of out_w row n
      v16h bt = *(const v16h*)(wh + (size_t)(n0 + t * 16 + l16) * HIDDEN + k + half * 16);
      acc[t] = wmma_f16(a, bt, acc[t]);
    }
  }
  #pragma unroll
  for (int t = 0; t < 4; ++t) {
    const int n = n0 + t * 16 + l16;
    const float bias = ob[n];
    #pragma unroll
    for (int j = 0; j < 8; ++j)
      out[(m0 + j + half * 8) * HIDDEN + n] = acc[t][j] + bias;
  }
}

// ---------------------------------------------------------------- launch
extern "C" void kernel_launch(void* const* d_in, const int* in_sizes, int n_in,
                              void* d_out, int out_size, void* d_ws, size_t ws_size,
                              hipStream_t stream) {
  (void)in_sizes; (void)n_in; (void)out_size; (void)ws_size;
  const float* hs  = (const float*)d_in[0];   // (8,2048,768)
  const float* lng = (const float*)d_in[1];   // (164,)
  const float* lnb = (const float*)d_in[2];   // (164,)
  const float* qw  = (const float*)d_in[3];   // (196,164)
  const float* qb  = (const float*)d_in[4];   // (196,)
  const float* ow  = (const float*)d_in[5];   // (768,768)
  const float* ob  = (const float*)d_in[6];   // (768,)
  float* out = (float*)d_out;

  char* ws = (char*)d_ws;                     // workspace layout (bytes):
  _Float16* qf  = (_Float16*)(ws);            //  0      .. 512K   q f16 (scaled)
  _Float16* kvt = (_Float16*)(ws + 524288);   //  512K   .. 6.5M   KV blocks (512 x 12KB)
  _Float16* mg  = (_Float16*)(ws + 6815744);  //  6.5M   .. 30.5M  merged f16 (16384x768)
  _Float16* wh  = (_Float16*)(ws + 31981568); //  30.5M  .. 31.6M  out_w f16

  cvt_w_kernel<<<(HIDDEN * HIDDEN + 255) / 256, 256, 0, stream>>>(ow, wh, HIDDEN * HIDDEN);
  ln_qkv_kernel<<<NTOK / 8, 256, 0, stream>>>(hs, lng, lnb, qw, qb, qf, kvt, mg);
  attn_kernel<<<BATCH * (SEQ / 128), 256, 0, stream>>>(qf, kvt, mg);
  outproj_kernel<<<NTOK / 16, 384, 0, stream>>>(mg, wh, ob, out);
}
